// FusedRobertaLayer_80908593922048
// MI455X (gfx1250) — matrix operand order repaired and checked by reading, not compiled, not run on hardware
//
#include <hip/hip_runtime.h>

// ---------------------------------------------------------------------------
// Fused RoBERTa layer with LoRA for MI455X (gfx1250, wave32, WMMA).
// LoRA folded into base weights (W' = W^T + la@lb) => pure GEMM pipeline:
// 3x GEMM -> flash attention -> GEMM(+resid) -> LN -> GEMM(+GELU)
// -> GEMM(+resid) -> LN.
// GEMM: v_wmma_f32_16x16x32_bf16, double-buffered LDS tiles staged by
//   - per-lane GLOBAL_LOAD_ASYNC_TO_LDS_B128 (ASYNCcnt) for activations
//   - a TDM TENSOR_LOAD_TO_LDS 2D descriptor (TENSORcnt) for weights,
// with depth-1 pipelining: s_wait_asynccnt 2 / s_wait_tensorcnt 1.
// ---------------------------------------------------------------------------

#define D_MODEL 1024
#define FF_DIM  4096
#define SEQ     512
#define BATCH   8
#define NHEAD   16
#define HDIM    64
#define MTOT    (BATCH * SEQ)   // 4096 token rows
#define LN_EPS  1e-5f

typedef __attribute__((ext_vector_type(16))) __bf16 v16bf;
typedef __attribute__((ext_vector_type(8)))  __bf16 v8bf;
typedef __attribute__((ext_vector_type(8)))  float  v8f;
typedef __attribute__((ext_vector_type(4)))  unsigned int u32x4;
typedef __attribute__((ext_vector_type(8)))  unsigned int u32x8;

union ABFrag { v16bf v; v8bf h[2]; };

// A-operand (16x32 bf16, M x K): lane r=lane&15 holds row r; half=lane>>4
// selects K groups {half*8 .. half*8+7} and {half*8+16 .. half*8+23}.
__device__ __forceinline__ ABFrag load_fragA(const __bf16* base, int stride,
                                             int row0, int k0) {
  int lane = threadIdx.x & 31;
  int r = lane & 15, half = lane >> 4;
  const __bf16* p = base + (size_t)(row0 + r) * stride + k0 + half * 8;
  ABFrag f;
  f.h[0] = *(const v8bf*)(p);
  f.h[1] = *(const v8bf*)(p + 16);
  return f;
}

// B-operand (32x16 bf16, K x N) loaded from a row-major [N,K] ("Bt") buffer:
// lane r holds column n=row0+r; half selects K groups {half*8..+7} and {half*8+16..+23}.
__device__ __forceinline__ ABFrag load_fragB(const __bf16* base, int stride,
                                             int row0, int k0) {
  int lane = threadIdx.x & 31;
  int r = lane & 15, half = lane >> 4;
  const __bf16* p = base + (size_t)(row0 + r) * stride + k0 + half * 8;
  ABFrag f;
  f.h[0] = *(const v8bf*)(p);
  f.h[1] = *(const v8bf*)(p + 16);
  return f;
}

__device__ __forceinline__ v8f wmma_bf16(const ABFrag& a, const ABFrag& b, v8f c) {
  return __builtin_amdgcn_wmma_f32_16x16x32_bf16(
      /*neg_a=*/false, a.v, /*neg_b=*/false, b.v,
      /*c_mod=*/(short)0, c, /*reuse_a=*/false, /*reuse_b=*/false);
}

// Async DMA 16 bytes global -> LDS (VFLAT GLOBAL_LOAD_ASYNC_TO_LDS_B128).
// LDS destination = low 32 bits of the generic pointer (flat aperture rule).
// Tracked by ASYNCcnt; loads retire in order.
__device__ __forceinline__ void async_load_b128(const __bf16* gptr, __bf16* lptr) {
  unsigned int  loff = (unsigned int)(unsigned long long)(uintptr_t)lptr;
  unsigned long long ga = (unsigned long long)(uintptr_t)gptr;
  asm volatile("global_load_async_to_lds_b128 %0, %1, off"
               :: "v"(loff), "v"(ga) : "memory");
}

__device__ __forceinline__ void wait_async_le0() {
  asm volatile("s_wait_asynccnt 0" ::: "memory");
}
__device__ __forceinline__ void wait_async_le2() {
  asm volatile("s_wait_asynccnt 2" ::: "memory");
}

// TDM: 2-D tile load, global -> LDS.  D# group0 (128b) + group1 (256b) built
// in SGPRs; tile_dim0=32 elems x tile_dim1=128 rows of a [rowsN x K] bf16
// tensor, with hardware LDS padding: every 16 DWORDs (64B row) insert
// 4 DWORDs (16B) => 80B LDS row stride == LDS_P.
__device__ __forceinline__ void tdm_load_tile_128x32(const __bf16* gtile,
                                                     __bf16* lds_dst, int K) {
  unsigned long long ga  = (unsigned long long)(uintptr_t)gtile;
  unsigned int       lds = (unsigned int)(unsigned long long)(uintptr_t)lds_dst;
  u32x4 g0;
  g0[0] = 1u;                                   // count=1, user descriptor
  g0[1] = lds;                                  // lds_addr
  g0[2] = (unsigned int)ga;                     // global_addr[31:0]
  g0[3] = (unsigned int)((ga >> 32) & 0x01FFFFFFu) | (2u << 30);  // addr[56:32] | type=2
  unsigned long long flags = (1ull << 16)       // data_size = 2 bytes
                           | (1ull << 20)       // pad_enable
                           | (3ull << 22)       // pad_interval: 16 DWORDs
                           | (3ull << 25);      // pad_amount:   4 DWORDs
  unsigned long long td0 = (unsigned long long)K;       // tensor_dim0
  unsigned long long td1 = 1ull << 20;                  // tensor_dim1 (generous)
  unsigned long long s0  = (unsigned long long)K;       // tensor_dim0_stride
  unsigned long long q0 = flags | (td0 << 48);
  unsigned long long q1 = (td0 >> 16) | (td1 << 16) | (32ull << 48);   // tile_dim0=32
  unsigned long long q2 = 128ull | (s0 << 32);                         // tile_dim1=128, tile_dim2=0
  unsigned long long q3 = (s0 >> 32);                                  // dim1_stride=0
  u32x8 g1;
  g1[0] = (unsigned int)q0; g1[1] = (unsigned int)(q0 >> 32);
  g1[2] = (unsigned int)q1; g1[3] = (unsigned int)(q1 >> 32);
  g1[4] = (unsigned int)q2; g1[5] = (unsigned int)(q2 >> 32);
  g1[6] = (unsigned int)q3; g1[7] = (unsigned int)(q3 >> 32);
  asm volatile("tensor_load_to_lds %0, %1" :: "s"(g0), "s"(g1) : "memory");
}

// ---------------------------------------------------------------------------
// fp32 -> bf16 cast
// ---------------------------------------------------------------------------
__global__ void cast_f32_bf16_kernel(const float* __restrict__ in,
                                     __bf16* __restrict__ out, long n) {
  long i = (long)blockIdx.x * blockDim.x + threadIdx.x;
  if (i < n) out[i] = (__bf16)in[i];
}

// ---------------------------------------------------------------------------
// LoRA fold: Wf[n][k] = w[n][k] + sum_r la[k][r] * lb[r][n]   (rank R=16)
// ---------------------------------------------------------------------------
__global__ void fuse_weight_kernel(const float* __restrict__ w,
                                   const float* __restrict__ la,
                                   const float* __restrict__ lb,
                                   __bf16* __restrict__ out, int N, int K) {
  long idx = (long)blockIdx.x * blockDim.x + threadIdx.x;
  if (idx >= (long)N * K) return;
  int k = (int)(idx % K);
  int n = (int)(idx / K);
  float acc = w[idx];
#pragma unroll
  for (int r = 0; r < 16; r++)
    acc += la[(size_t)k * 16 + r] * lb[(size_t)r * N + n];
  out[idx] = (__bf16)acc;
}

// ---------------------------------------------------------------------------
// Tiled WMMA GEMM: C[M,N] = A[M,K](bf16) @ Bt[N,K](bf16)^T + bias (+resid)
// Block tile 128x128, K-step 32, double-buffered LDS.
// 8 waves as 4(M) x 2(N); wave tile 32x64: 2 A-frags x 4 B-frags = 8 WMMA/step.
// A tile: per-lane async DMA (2 b128/thread).  Bt tile: one TDM descriptor.
// ---------------------------------------------------------------------------
#define GBK   32
#define LDS_P 40   // padded row stride (bf16 elems); 80B keeps 16B alignment

__device__ __forceinline__ void stage_tiles(const __bf16* A, const __bf16* Bt,
                                            __bf16 (*sA)[LDS_P], __bf16 (*sB)[LDS_P],
                                            int m_blk, int n_blk, int K, int k0,
                                            int tid, int wid) {
#pragma unroll
  for (int c = tid; c < 512; c += 256) {       // A tile: 128x32, 2 b128/thread
    int row = c >> 2, ko = (c & 3) * 8;
    async_load_b128(A + (size_t)(m_blk + row) * K + k0 + ko, &sA[row][ko]);
  }
  if (wid == 0)                                 // Bt tile: one TDM per workgroup
    tdm_load_tile_128x32(Bt + (size_t)n_blk * K + k0, &sB[0][0], K);
}

template <bool OUT_BF16, bool GELU>
__global__ __launch_bounds__(256)
void gemm_bias_kernel(const __bf16* __restrict__ A, const __bf16* __restrict__ Bt,
                      const float* __restrict__ bias, const float* __restrict__ resid,
                      float* __restrict__ Cf, __bf16* __restrict__ Cb,
                      int M, int N, int K) {
  __shared__ __align__(16) __bf16 As[2][128][LDS_P];
  __shared__ __align__(16) __bf16 Bs[2][128][LDS_P];

  int tid = threadIdx.x;
  int wid = tid >> 5;
  int wm = wid & 3, wn = wid >> 2;
  int lane = tid & 31, half = lane >> 4;
  int m_blk = blockIdx.x * 128;
  int n_blk = blockIdx.y * 128;
  int nk = K / GBK;

  v8f acc[2][4] = {};

  stage_tiles(A, Bt, As[0], Bs[0], m_blk, n_blk, K, 0, tid, wid);

  for (int kt = 0; kt < nk; kt++) {
    int cur = kt & 1;
    if (kt + 1 < nk) {
      // prefetch next tile into the other buffer (safe: end-of-iter barrier
      // of kt-1 guaranteed all reads of that buffer completed)
      stage_tiles(A, Bt, As[cur ^ 1], Bs[cur ^ 1], m_blk, n_blk, K,
                  (kt + 1) * GBK, tid, wid);
      wait_async_le2();                         // cur's 2 loads/thread done
      if (wid == 0) __builtin_amdgcn_s_wait_tensorcnt(1);
    } else {
      wait_async_le0();
      if (wid == 0) __builtin_amdgcn_s_wait_tensorcnt(0);
    }
    __syncthreads();                            // cur buffer visible to all

    ABFrag a0 = load_fragA(&As[cur][wm * 32][0],      LDS_P, 0, 0);
    ABFrag a1 = load_fragA(&As[cur][wm * 32 + 16][0], LDS_P, 0, 0);
#pragma unroll
    for (int t = 0; t < 4; t++) {
      ABFrag b = load_fragB(&Bs[cur][wn * 64 + t * 16][0], LDS_P, 0, 0);
      acc[0][t] = wmma_bf16(a0, b, acc[0][t]);
      acc[1][t] = wmma_bf16(a1, b, acc[1][t]);
    }
    __syncthreads();                            // reads of cur done everywhere
  }

  // Epilogue: C-frag rows = e + 8*half, cols = lane&15 within 16-wide tile.
#pragma unroll
  for (int u = 0; u < 2; u++) {
#pragma unroll
    for (int t = 0; t < 4; t++) {
      int col = n_blk + wn * 64 + t * 16 + (lane & 15);
      float bv = bias[col];
#pragma unroll
      for (int e = 0; e < 8; e++) {
        int row = m_blk + wm * 32 + u * 16 + e + 8 * half;
        float v = acc[u][t][e] + bv;
        if (resid) v += resid[(size_t)row * N + col];
        if (GELU) v = 0.5f * v * (1.0f + erff(v * 0.70710678118654752f));
        if (OUT_BF16) Cb[(size_t)row * N + col] = (__bf16)v;
        else          Cf[(size_t)row * N + col] = v;
      }
    }
  }
}

// ---------------------------------------------------------------------------
// Flash attention: grid (S/128, H, B), 8 waves x 16 q-rows each.
// K/V staged per 64-key chunk in LDS (V transposed), online softmax in
// C-fragment registers, P transposed to A-layout via a per-wave LDS bounce.
// ---------------------------------------------------------------------------
#define ALD 72  // padded LDS row stride (bf16); 144B = 16B multiple

__global__ __launch_bounds__(256)
void attention_kernel(const __bf16* __restrict__ Qb, const __bf16* __restrict__ Kb,
                      const __bf16* __restrict__ Vb, const float* __restrict__ mask,
                      __bf16* __restrict__ Ob) {
  __shared__ __align__(16) __bf16 Ks[64][ALD];
  __shared__ __align__(16) __bf16 Vt[64][ALD];
  __shared__ __align__(16) __bf16 Pb[8][16][ALD];

  int tid = threadIdx.x, wid = tid >> 5, lane = tid & 31, half = lane >> 4;
  int qt = blockIdx.x, h = blockIdx.y, b = blockIdx.z;
  int qrow0 = b * SEQ + qt * 128 + wid * 16;  // row into [MTOT, D_MODEL]
  int colh = h * HDIM;

  ABFrag q0 = load_fragA(Qb + colh, D_MODEL, qrow0, 0);
  ABFrag q1 = load_fragA(Qb + colh, D_MODEL, qrow0, 32);

  float mrow[8], lrow[8];
  v8f oacc[4] = {{}, {}, {}, {}};
#pragma unroll
  for (int e = 0; e < 8; e++) { mrow[e] = -1e30f; lrow[e] = 0.f; }

  for (int j = 0; j < SEQ / 64; j++) {
    __syncthreads();
    for (int c = tid; c < 512; c += 256) {  // stage 64x64 K (row) + V (transposed)
      int r = c >> 3, ko = (c & 7) * 8;
      size_t g = (size_t)(b * SEQ + j * 64 + r) * D_MODEL + colh + ko;
      *(v8bf*)&Ks[r][ko] = *(const v8bf*)(Kb + g);
      v8bf vv = *(const v8bf*)(Vb + g);
#pragma unroll
      for (int i = 0; i < 8; i++) Vt[ko + i][r] = vv[i];
    }
    __syncthreads();

    // scores S = Q K^T  (16 q-rows x 64 keys, HD=64 => 2 wmma per 16-key tile)
    v8f s[4];
#pragma unroll
    for (int t = 0; t < 4; t++) {
      v8f z = {};
      ABFrag kf0 = load_fragB(&Ks[t * 16][0], ALD, 0, 0);
      ABFrag kf1 = load_fragB(&Ks[t * 16][0], ALD, 0, 32);
      z = wmma_bf16(q0, kf0, z);
      z = wmma_bf16(q1, kf1, z);
      s[t] = z;
    }
#pragma unroll
    for (int t = 0; t < 4; t++) {  // 1/sqrt(64) scale + additive mask
      float mv = mask[b * SEQ + j * 64 + t * 16 + (lane & 15)];
#pragma unroll
      for (int e = 0; e < 8; e++) s[t][e] = s[t][e] * 0.125f + mv;
    }

    // online softmax; row of element e is (e + 8*half); row reductions are
    // across the 16-lane half (shfl_xor masks 1/2/4/8 stay within it)
#pragma unroll
    for (int e = 0; e < 8; e++) {
      float m = fmaxf(fmaxf(s[0][e], s[1][e]), fmaxf(s[2][e], s[3][e]));
#pragma unroll
      for (int off = 8; off >= 1; off >>= 1) m = fmaxf(m, __shfl_xor(m, off, 32));
      float mnew = fmaxf(mrow[e], m);
      float corr = __expf(mrow[e] - mnew);
      mrow[e] = mnew;
      lrow[e] *= corr;
#pragma unroll
      for (int t = 0; t < 4; t++) oacc[t][e] = oacc[t][e] * corr;
      float rsum = 0.f;
#pragma unroll
      for (int t = 0; t < 4; t++) {
        float p = __expf(s[t][e] - mnew);
        s[t][e] = p;
        rsum += p;
      }
#pragma unroll
      for (int off = 8; off >= 1; off >>= 1) rsum += __shfl_xor(rsum, off, 32);
      lrow[e] += rsum;
    }

    // C-layout -> A-layout transpose of P through per-wave LDS bounce
#pragma unroll
    for (int t = 0; t < 4; t++)
#pragma unroll
      for (int e = 0; e < 8; e++)
        Pb[wid][e + 8 * half][t * 16 + (lane & 15)] = (__bf16)s[t][e];

    ABFrag p0 = load_fragA(&Pb[wid][0][0], ALD, 0, 0);
    ABFrag p1 = load_fragA(&Pb[wid][0][0], ALD, 0, 32);
#pragma unroll
    for (int t = 0; t < 4; t++) {  // O += P @ V  (Vt rows are output columns)
      ABFrag vf0 = load_fragB(&Vt[t * 16][0], ALD, 0, 0);
      ABFrag vf1 = load_fragB(&Vt[t * 16][0], ALD, 0, 32);
      oacc[t] = wmma_bf16(p0, vf0, oacc[t]);
      oacc[t] = wmma_bf16(p1, vf1, oacc[t]);
    }
  }

#pragma unroll
  for (int t = 0; t < 4; t++) {
    int col = colh + t * 16 + (lane & 15);
#pragma unroll
    for (int e = 0; e < 8; e++) {
      int row = qrow0 + e + 8 * half;
      Ob[(size_t)row * D_MODEL + col] = (__bf16)(oacc[t][e] / lrow[e]);
    }
  }
}

// ---------------------------------------------------------------------------
// LayerNorm: one 256-thread block per row; shfl + LDS reduction.
// ---------------------------------------------------------------------------
__global__ __launch_bounds__(256)
void ln_kernel(const float* __restrict__ in, const float* __restrict__ w,
               const float* __restrict__ bb, float* __restrict__ outf,
               __bf16* __restrict__ outb, int D) {
  int row = blockIdx.x;
  const float* x = in + (size_t)row * D;
  float s = 0.f, ss = 0.f;
  for (int c = threadIdx.x; c < D; c += 256) {
    float v = x[c];
    s += v; ss += v * v;
  }
#pragma unroll
  for (int off = 16; off >= 1; off >>= 1) {
    s += __shfl_xor(s, off, 32);
    ss += __shfl_xor(ss, off, 32);
  }
  __shared__ float rs[8], rss[8];
  int wid = threadIdx.x >> 5, lane = threadIdx.x & 31;
  if (lane == 0) { rs[wid] = s; rss[wid] = ss; }
  __syncthreads();
  s = 0.f; ss = 0.f;
#pragma unroll
  for (int i = 0; i < 8; i++) { s += rs[i]; ss += rss[i]; }
  float mu = s / D;
  float var = ss / D - mu * mu;
  float inv = rsqrtf(var + LN_EPS);
  for (int c = threadIdx.x; c < D; c += 256) {
    float v = (x[c] - mu) * inv * w[c] + bb[c];
    outf[(size_t)row * D + c] = v;
    if (outb) outb[(size_t)row * D + c] = (__bf16)v;
  }
}

// ---------------------------------------------------------------------------
extern "C" void kernel_launch(void* const* d_in, const int* in_sizes, int n_in,
                              void* d_out, int out_size, void* d_ws, size_t ws_size,
                              hipStream_t stream) {
  (void)in_sizes; (void)n_in; (void)out_size; (void)ws_size;
  const float* x     = (const float*)d_in[0];
  const float* amask = (const float*)d_in[1];
  const float* w_q = (const float*)d_in[2];  const float* b_q = (const float*)d_in[3];
  const float* la_q = (const float*)d_in[4]; const float* lb_q = (const float*)d_in[5];
  const float* w_k = (const float*)d_in[6];  const float* b_k = (const float*)d_in[7];
  const float* la_k = (const float*)d_in[8]; const float* lb_k = (const float*)d_in[9];
  const float* w_v = (const float*)d_in[10]; const float* b_v = (const float*)d_in[11];
  const float* la_v = (const float*)d_in[12]; const float* lb_v = (const float*)d_in[13];
  const float* w_o = (const float*)d_in[14]; const float* b_o = (const float*)d_in[15];
  const float* la_o = (const float*)d_in[16]; const float* lb_o = (const float*)d_in[17];
  const float* nw1 = (const float*)d_in[18]; const float* nb1 = (const float*)d_in[19];
  const float* w_up = (const float*)d_in[20]; const float* b_up = (const float*)d_in[21];
  const float* la_up = (const float*)d_in[22]; const float* lb_up = (const float*)d_in[23];
  const float* w_dn = (const float*)d_in[24]; const float* b_dn = (const float*)d_in[25];
  const float* la_dn = (const float*)d_in[26]; const float* lb_dn = (const float*)d_in[27];
  const float* nw2 = (const float*)d_in[28]; const float* nb2 = (const float*)d_in[29];

  char* ws = (char*)d_ws;
  size_t off = 0;
  auto alloc = [&](size_t bytes) -> void* {
    size_t cur = (off + 255) & ~(size_t)255;
    off = cur + bytes;
    return (void*)(ws + cur);
  };

  __bf16* Xb   = (__bf16*)alloc((size_t)MTOT * D_MODEL * 2);
  __bf16* Wfq  = (__bf16*)alloc((size_t)D_MODEL * D_MODEL * 2);
  __bf16* Wfk  = (__bf16*)alloc((size_t)D_MODEL * D_MODEL * 2);
  __bf16* Wfv  = (__bf16*)alloc((size_t)D_MODEL * D_MODEL * 2);
  __bf16* Wfo  = (__bf16*)alloc((size_t)D_MODEL * D_MODEL * 2);
  __bf16* Wfup = (__bf16*)alloc((size_t)FF_DIM * D_MODEL * 2);
  __bf16* Wfdn = (__bf16*)alloc((size_t)D_MODEL * FF_DIM * 2);
  __bf16* qb   = (__bf16*)alloc((size_t)MTOT * D_MODEL * 2);
  __bf16* kb   = (__bf16*)alloc((size_t)MTOT * D_MODEL * 2);
  __bf16* vb   = (__bf16*)alloc((size_t)MTOT * D_MODEL * 2);
  __bf16* ob   = (__bf16*)alloc((size_t)MTOT * D_MODEL * 2);
  float*  ofin = (float*) alloc((size_t)MTOT * D_MODEL * 4);
  float*  xm   = (float*) alloc((size_t)MTOT * D_MODEL * 4);
  __bf16* xmb  = (__bf16*)alloc((size_t)MTOT * D_MODEL * 2);
  __bf16* fnb  = (__bf16*)alloc((size_t)MTOT * FF_DIM * 2);
  float*  dwn  = (float*) alloc((size_t)MTOT * D_MODEL * 4);

  long nX = (long)MTOT * D_MODEL;
  cast_f32_bf16_kernel<<<(nX + 255) / 256, 256, 0, stream>>>(x, Xb, nX);

  long nDD = (long)D_MODEL * D_MODEL, nFD = (long)FF_DIM * D_MODEL;
  fuse_weight_kernel<<<(nDD + 255) / 256, 256, 0, stream>>>(w_q, la_q, lb_q, Wfq, D_MODEL, D_MODEL);
  fuse_weight_kernel<<<(nDD + 255) / 256, 256, 0, stream>>>(w_k, la_k, lb_k, Wfk, D_MODEL, D_MODEL);
  fuse_weight_kernel<<<(nDD + 255) / 256, 256, 0, stream>>>(w_v, la_v, lb_v, Wfv, D_MODEL, D_MODEL);
  fuse_weight_kernel<<<(nDD + 255) / 256, 256, 0, stream>>>(w_o, la_o, lb_o, Wfo, D_MODEL, D_MODEL);
  fuse_weight_kernel<<<(nFD + 255) / 256, 256, 0, stream>>>(w_up, la_up, lb_up, Wfup, FF_DIM, D_MODEL);
  fuse_weight_kernel<<<(nFD + 255) / 256, 256, 0, stream>>>(w_dn, la_dn, lb_dn, Wfdn, D_MODEL, FF_DIM);

  dim3 gD(MTOT / 128, D_MODEL / 128);
  dim3 gF(MTOT / 128, FF_DIM / 128);

  // Q, K, V projections (bf16 out, feeding attention)
  gemm_bias_kernel<true, false><<<gD, 256, 0, stream>>>(Xb, Wfq, b_q, nullptr, nullptr, qb, MTOT, D_MODEL, D_MODEL);
  gemm_bias_kernel<true, false><<<gD, 256, 0, stream>>>(Xb, Wfk, b_k, nullptr, nullptr, kb, MTOT, D_MODEL, D_MODEL);
  gemm_bias_kernel<true, false><<<gD, 256, 0, stream>>>(Xb, Wfv, b_v, nullptr, nullptr, vb, MTOT, D_MODEL, D_MODEL);

  attention_kernel<<<dim3(SEQ / 128, NHEAD, BATCH), 256, 0, stream>>>(qb, kb, vb, amask, ob);

  // O projection + residual x
  gemm_bias_kernel<false, false><<<gD, 256, 0, stream>>>(ob, Wfo, b_o, x, ofin, nullptr, MTOT, D_MODEL, D_MODEL);
  ln_kernel<<<MTOT, 256, 0, stream>>>(ofin, nw1, nb1, xm, xmb, D_MODEL);

  // FFN up + exact GELU (bf16 out), down + residual x_medium
  gemm_bias_kernel<true, true><<<gF, 256, 0, stream>>>(xmb, Wfup, b_up, nullptr, nullptr, fnb, MTOT, FF_DIM, D_MODEL);
  gemm_bias_kernel<false, false><<<gD, 256, 0, stream>>>(fnb, Wfdn, b_dn, xm, dwn, nullptr, MTOT, D_MODEL, FF_DIM);

  ln_kernel<<<MTOT, 256, 0, stream>>>(dwn, nw2, nb2, (float*)d_out, nullptr, D_MODEL);
}
